// CrossAttention_54992761258105
// MI455X (gfx1250) — matrix operand order repaired and checked by reading, not compiled, not run on hardware
//
#include <hip/hip_runtime.h>
#include <hip/hip_bf16.h>

// ---------------------------------------------------------------------------
// CrossAttention for MI455X (gfx1250, wave32, WMMA).
//   B=8, LQ=LK=1024, QD=512, KD=VD=256, OD=512, H=8, HD=64, SCALE=8
// d_out = [ out: B*LQ*OD f32 | attention: B*H*LQ*LK f32 ]
// d_ws  = [ q16 | k16 | v16 | oh16 ]  (f16, 8 MB each = 32 MB total)
// All GEMMs use v_wmma_f32_16x16x32_f16 (f16 inputs, f32 accumulate).
// Final projection stages its f16 A tiles with GLOBAL_LOAD_ASYNC_TO_LDS_B128
// (ASYNCcnt / s_wait_asynccnt) when the toolchain exposes the builtin.
// ---------------------------------------------------------------------------

#define B_   8
#define LQ_  1024
#define LK_  1024
#define QD_  512
#define KD_  256
#define VD_  256
#define OD_  512
#define H_   8
#define HD_  64
#define INV_SCALE 0.125f

typedef _Float16 h16;
typedef __attribute__((ext_vector_type(16))) _Float16 v16h;
typedef __attribute__((ext_vector_type(8)))  _Float16 v8h;
typedef __attribute__((ext_vector_type(8)))  float    v8f;
typedef __attribute__((ext_vector_type(4)))  int      i32x4;

#if defined(__has_builtin)
# if __has_builtin(__builtin_amdgcn_global_load_async_to_lds_b128)
#  define ASYNC_LDS 1
# endif
#endif
#ifndef ASYNC_LDS
# define ASYNC_LDS 0
#endif

__device__ __forceinline__ void async_wait0() {
#if ASYNC_LDS
# if __has_builtin(__builtin_amdgcn_s_wait_asynccnt)
  __builtin_amdgcn_s_wait_asynccnt(0);
# else
  asm volatile("s_wait_asynccnt 0" ::: "memory");
# endif
#endif
}

union V16U { v16h v; v8h h[2]; };

// A-fragment (16x32 f16): lane%16 = row, k = (lane>>4)*16 + e (contiguous).
// p points at row base + (lane>>4)*16.
__device__ __forceinline__ v16h frag_a(const h16* p) {
  V16U u;
  u.h[0] = *(const v8h*)(p);
  u.h[1] = *(const v8h*)(p + 8);
  return u.v;
}
// B-fragment (32x16 f16): lane%16 = column, k = (lane>>4)*16 + e (contiguous).
// p points at column base + (lane>>4)*16 in a [n][k]-major buffer.
__device__ __forceinline__ v16h frag_b(const h16* p) {
  V16U u;
  u.h[0] = *(const v8h*)(p);
  u.h[1] = *(const v8h*)(p + 8);
  return u.v;
}
__device__ __forceinline__ v8f wmma32(v16h a, v16h b, v8f c) {
  return __builtin_amdgcn_wmma_f32_16x16x32_f16(
      /*neg_a=*/false, a, /*neg_b=*/false, b,
      /*c_mod=*/(short)0, c, /*reuse_a=*/false, /*reuse_b=*/false);
}

// Stage 16 halves (one A-tile slice) into LDS.
// f32 source: convert elementwise.  f16 source: async global->LDS copy.
__device__ __forceinline__ void stage_a16(const float* g, h16* d) {
#pragma unroll
  for (int i = 0; i < 16; ++i) d[i] = (h16)g[i];
}
__device__ __forceinline__ void stage_a16(const h16* g, h16* d) {
#if ASYNC_LDS
  __builtin_amdgcn_global_load_async_to_lds_b128(
      (__attribute__((address_space(1))) i32x4*)(g),
      (__attribute__((address_space(3))) i32x4*)(d), 0, 0);
  __builtin_amdgcn_global_load_async_to_lds_b128(
      (__attribute__((address_space(1))) i32x4*)(g + 8),
      (__attribute__((address_space(3))) i32x4*)(d + 8), 0, 0);
#else
#pragma unroll
  for (int i = 0; i < 16; ++i) d[i] = g[i];
#endif
}

// ---------------------------------------------------------------------------
// Generic projection GEMM: out(M,N) = A(M,K) @ W(K,N) + bias(N)
// Block tile 128x64, 256 threads (8 waves, 4x2), each wave 32x32 (2x2 WMMA).
// LDS: A row-major [m][k] stride 48 halves; W transposed [n][k] stride 48.
// ---------------------------------------------------------------------------
template <typename AT, typename OT>
__global__ __launch_bounds__(256) void proj_gemm(
    const AT* __restrict__ A, const float* __restrict__ W,
    const float* __restrict__ bias, OT* __restrict__ out,
    int M, int K, int N) {
  __shared__ h16 As[128 * 48];
  __shared__ h16 Ws[64 * 48];
  const int tid  = threadIdx.x;
  const int lane = tid & 31;
  const int wid  = tid >> 5;
  const int wm   = wid >> 1;          // 0..3
  const int wn   = wid & 1;           // 0..1
  const int m0   = blockIdx.x * 128;
  const int n0   = blockIdx.y * 64;

  const int arow = tid >> 1, ac0 = (tid & 1) * 16;       // A-tile mapping
  const int wk = tid >> 3, wc0 = (tid & 7) * 8;          // W-tile mapping
  const AT*    gA = A + (size_t)(m0 + arow) * K + ac0;
  const float* gW = W + (size_t)wk * N + n0 + wc0;

  v8f acc[2][2] = {};
  for (int k0 = 0; k0 < K; k0 += 32) {
    stage_a16(gA + k0, As + arow * 48 + ac0);            // A tile 128x32
    {                                                    // W tile 32x64 -> [n][k]
      const float* g = gW + (size_t)k0 * N;
#pragma unroll
      for (int i = 0; i < 8; ++i) Ws[(wc0 + i) * 48 + wk] = (h16)g[i];
    }
    if (k0 + 32 < K) {                                   // prefetch next k-step
      __builtin_prefetch(gA + k0 + 32, 0, 0);
      __builtin_prefetch(gW + (size_t)(k0 + 32) * N, 0, 0);
    }
    async_wait0();
    __syncthreads();
#pragma unroll
    for (int mt = 0; mt < 2; ++mt) {
      v16h a = frag_a(As + (wm * 32 + mt * 16 + (lane & 15)) * 48 + ((lane >> 4) * 16));
#pragma unroll
      for (int nt = 0; nt < 2; ++nt) {
        v16h b = frag_b(Ws + (wn * 32 + nt * 16 + (lane & 15)) * 48 + ((lane >> 4) * 16));
        acc[mt][nt] = wmma32(a, b, acc[mt][nt]);
      }
    }
    __syncthreads();
  }
#pragma unroll
  for (int mt = 0; mt < 2; ++mt)
#pragma unroll
    for (int nt = 0; nt < 2; ++nt) {
      const int gcol = n0 + wn * 32 + nt * 16 + (lane & 15);
      const float bv = bias[gcol];
      const int rb = m0 + wm * 32 + mt * 16 + ((lane >> 4) * 8);
#pragma unroll
      for (int r = 0; r < 8; ++r)
        out[(size_t)(rb + r) * N + gcol] = (OT)(acc[mt][nt][r] + bv);
    }
}

// ---------------------------------------------------------------------------
// scores = softmax(Q_h K_h^T / 8), written to d_out attention region (f32).
// One block = (b, h, 16-row q-tile). 128 threads = 4 waves; wave w owns
// columns [w*256, w*256+256). Full 16x1024 score stripe lives in 64 KB LDS.
// Two n-tiles processed per iteration with independent accumulators so the
// scheduler can hide the F16 WMMA->WMMA RAW hazard (ISA 7.12.1).
// ---------------------------------------------------------------------------
__global__ __launch_bounds__(128) void attn_scores_softmax(
    const h16* __restrict__ q16, const h16* __restrict__ k16,
    float* __restrict__ attn) {
  __shared__ float sc[16 * 1024];               // 64 KB
  const int t   = blockIdx.x;                   // (b, h, qt)
  const int qt  = t & 63;                       // LQ/16 = 64
  const int h   = (t >> 6) & 7;
  const int b   = t >> 9;
  const int tid = threadIdx.x;
  const int lane = tid & 31;
  const int w    = tid >> 5;
  const int q0   = qt * 16;

  // Q fragments (16 rows x 64 hd) = two 16x32 A-frags, loaded straight from
  // global f16 (runs of 8 halves are 16B-aligned).
  const h16* qb = q16 + ((size_t)(b * LQ_ + q0 + (lane & 15))) * OD_ +
                  h * HD_ + ((lane >> 4) * 16);
  const v16h a0 = frag_a(qb);
  const v16h a1 = frag_a(qb + 32);

  const h16* kbase = k16 + ((size_t)(b * LK_ + (lane & 15))) * OD_ +
                     h * HD_ + ((lane >> 4) * 16);
  const int sr = (lane >> 4) * 8;

  for (int nt = 0; nt < 16; nt += 2) {
    const int cb0 = w * 256 + nt * 16;          // global key-position bases
    const int cb1 = cb0 + 16;
    const h16* kp0 = kbase + (size_t)cb0 * OD_;
    const h16* kp1 = kbase + (size_t)cb1 * OD_;
    const v16h b00 = frag_b(kp0);
    const v16h b01 = frag_b(kp0 + 32);
    const v16h b10 = frag_b(kp1);
    const v16h b11 = frag_b(kp1 + 32);
    v8f c0 = {}, c1 = {};
    c0 = wmma32(a0, b00, c0);
    c1 = wmma32(a0, b10, c1);                   // independent chain interleaves
    c0 = wmma32(a1, b01, c0);
    c1 = wmma32(a1, b11, c1);
    const int s0 = cb0 + (lane & 15);
    const int s1 = cb1 + (lane & 15);
#pragma unroll
    for (int r = 0; r < 8; ++r) {
      sc[(sr + r) * 1024 + s0] = c0[r] * INV_SCALE;
      sc[(sr + r) * 1024 + s1] = c1[r] * INV_SCALE;
    }
  }
  __syncthreads();

  // Softmax: 8 threads per row (128 cols each); wave32 shfl_xor reductions.
  const int row = tid >> 3, sub = tid & 7;
  float* sp = sc + row * 1024 + sub * 128;
  float m = -3.4e38f;
  for (int j = 0; j < 128; j += 4) {
    const float4 v = *(const float4*)(sp + j);
    m = fmaxf(m, fmaxf(fmaxf(v.x, v.y), fmaxf(v.z, v.w)));
  }
  m = fmaxf(m, __shfl_xor(m, 1, 8));
  m = fmaxf(m, __shfl_xor(m, 2, 8));
  m = fmaxf(m, __shfl_xor(m, 4, 8));
  float s = 0.0f;
  for (int j = 0; j < 128; j += 4) {
    float4 v = *(float4*)(sp + j);
    v.x = __expf(v.x - m); v.y = __expf(v.y - m);
    v.z = __expf(v.z - m); v.w = __expf(v.w - m);
    *(float4*)(sp + j) = v;
    s += v.x + v.y + v.z + v.w;
  }
  s += __shfl_xor(s, 1, 8);
  s += __shfl_xor(s, 2, 8);
  s += __shfl_xor(s, 4, 8);
  const float inv = 1.0f / s;
  float* g = attn + ((size_t)((b * H_ + h) * LQ_ + q0 + row)) * LK_ + sub * 128;
  for (int j = 0; j < 128; j += 4) {
    float4 v = *(const float4*)(sp + j);
    v.x *= inv; v.y *= inv; v.z *= inv; v.w *= inv;
    *(float4*)(g + j) = v;
  }
}

// ---------------------------------------------------------------------------
// out_heads = P @ V_h : per (b,h), (1024x1024) @ (1024x64).
// Block tile 64x64, 128 threads (4 waves, 2x2), K-loop over LK in steps of 32.
// P read as f32 from d_out attention region, converted to f16 in LDS.
// ---------------------------------------------------------------------------
__global__ __launch_bounds__(128) void attn_pv(
    const float* __restrict__ attn, const h16* __restrict__ vh,
    h16* __restrict__ oh) {
  __shared__ h16 Ps[64 * 48];
  __shared__ h16 Vs[64 * 48];
  const int t   = blockIdx.x;                   // (b, h, 64-row q-tile)
  const int qt  = t & 15;                       // LQ/64 = 16
  const int h   = (t >> 4) & 7;
  const int b   = t >> 7;
  const int q0  = qt * 64;
  const int tid = threadIdx.x;
  const int lane = tid & 31;
  const int wid  = tid >> 5;
  const int wm = wid >> 1, wn = wid & 1;

  const int prow = tid >> 1, pc0 = (tid & 1) * 16;
  const int vkk = tid >> 2, vd0 = (tid & 3) * 16;
  const float* gP = attn +
      ((size_t)((b * H_ + h) * LQ_ + q0 + prow)) * LK_ + pc0;
  const h16* gV = vh + ((size_t)(b * LK_ + vkk)) * OD_ + h * HD_ + vd0;

  v8f acc[2][2] = {};
  for (int k0 = 0; k0 < LK_; k0 += 32) {
    {   // P tile 64x32 f32 -> f16 LDS row-major
      const float* g = gP + k0;
      h16* d = Ps + prow * 48 + pc0;
#pragma unroll
      for (int i = 0; i < 16; ++i) d[i] = (h16)g[i];
    }
    {   // V tile 32x64 f16 -> transposed LDS [d][k]
      const h16* g = gV + (size_t)k0 * OD_;
#pragma unroll
      for (int i = 0; i < 16; ++i) Vs[(vd0 + i) * 48 + vkk] = g[i];
    }
    __syncthreads();
#pragma unroll
    for (int mt = 0; mt < 2; ++mt) {
      v16h a = frag_a(Ps + (wm * 32 + mt * 16 + (lane & 15)) * 48 + ((lane >> 4) * 16));
#pragma unroll
      for (int nt = 0; nt < 2; ++nt) {
        v16h bf = frag_b(Vs + (wn * 32 + nt * 16 + (lane & 15)) * 48 + ((lane >> 4) * 16));
        acc[mt][nt] = wmma32(a, bf, acc[mt][nt]);
      }
    }
    __syncthreads();
  }
#pragma unroll
  for (int mt = 0; mt < 2; ++mt)
#pragma unroll
    for (int nt = 0; nt < 2; ++nt) {
      const int d  = wn * 32 + nt * 16 + (lane & 15);
      const int rb = q0 + wm * 32 + mt * 16 + ((lane >> 4) * 8);
#pragma unroll
      for (int r = 0; r < 8; ++r)
        oh[((size_t)(b * LQ_ + rb + r)) * OD_ + h * HD_ + d] = (h16)acc[mt][nt][r];
    }
}

// ---------------------------------------------------------------------------
extern "C" void kernel_launch(void* const* d_in, const int* in_sizes, int n_in,
                              void* d_out, int out_size, void* d_ws,
                              size_t ws_size, hipStream_t stream) {
  (void)in_sizes; (void)n_in; (void)out_size; (void)ws_size;
  const float* query = (const float*)d_in[0];
  const float* key_  = (const float*)d_in[1];
  const float* value = (const float*)d_in[2];
  const float* Wq = (const float*)d_in[3]; const float* bq = (const float*)d_in[4];
  const float* Wk = (const float*)d_in[5]; const float* bk = (const float*)d_in[6];
  const float* Wv = (const float*)d_in[7]; const float* bv = (const float*)d_in[8];
  const float* Wo = (const float*)d_in[9]; const float* bo = (const float*)d_in[10];

  float* out  = (float*)d_out;
  float* attn = out + (size_t)B_ * LQ_ * OD_;   // second tuple element

  const size_t act = (size_t)B_ * LQ_ * OD_;    // 4,194,304 halves each
  h16* q16  = (h16*)d_ws;
  h16* k16  = q16 + act;
  h16* v16p = k16 + act;
  h16* oh16 = v16p + act;                       // total 32 MB of d_ws

  const dim3 pblk(256);
  const dim3 pgrid(B_ * LQ_ / 128, OD_ / 64);
  proj_gemm<float, h16><<<pgrid, pblk, 0, stream>>>(query, Wq, bq, q16,
                                                    B_ * LQ_, QD_, OD_);
  proj_gemm<float, h16><<<pgrid, pblk, 0, stream>>>(key_, Wk, bk, k16,
                                                    B_ * LK_, KD_, OD_);
  proj_gemm<float, h16><<<pgrid, pblk, 0, stream>>>(value, Wv, bv, v16p,
                                                    B_ * LK_, VD_, OD_);
  attn_scores_softmax<<<dim3(B_ * H_ * (LQ_ / 16)), dim3(128), 0, stream>>>(
      q16, k16, attn);
  attn_pv<<<dim3(B_ * H_ * (LQ_ / 64)), dim3(128), 0, stream>>>(attn, v16p,
                                                                oh16);
  proj_gemm<h16, float><<<pgrid, pblk, 0, stream>>>(oh16, Wo, bo, out,
                                                    B_ * LQ_, OD_, OD_);
}
